// StyleHyperLinear_44684839747821
// MI455X (gfx1250) — matrix operand order repaired and checked by reading, not compiled, not run on hardware
//
#include <hip/hip_runtime.h>
#include <hip/hip_bf16.h>
#include <math.h>

// ---- types -----------------------------------------------------------------
typedef __bf16 bf16_t;
typedef __attribute__((ext_vector_type(16))) __bf16 v16bf;
typedef __attribute__((ext_vector_type(8)))  __bf16 v8bf;
typedef __attribute__((ext_vector_type(8)))  float  v8f;
typedef __attribute__((ext_vector_type(4)))  float  f32x4;
typedef __attribute__((ext_vector_type(4)))  unsigned int u32x4;
typedef __attribute__((ext_vector_type(4)))  int    i32x4;

// address-space-qualified pointee types for the async-to-LDS builtins
typedef __attribute__((address_space(1))) int   as1_i32;
typedef __attribute__((address_space(3))) int   as3_i32;
typedef __attribute__((address_space(1))) i32x4 as1_i32x4;
typedef __attribute__((address_space(3))) i32x4 as3_i32x4;

#define IN_F   1024
#define OUT_F  1024
#define STYLE  256
#define RANK   8
#define NBATCH 16
#define NSEQ   4096

#if defined(__HIP_DEVICE_COMPILE__) &&                                   \
    __has_builtin(__builtin_amdgcn_global_load_async_to_lds_b128) &&     \
    __has_builtin(__builtin_amdgcn_global_load_async_to_lds_b32)  &&     \
    __has_builtin(__builtin_amdgcn_s_wait_asynccnt)
#define ASYNC_OK 1
#else
#define ASYNC_OK 0
#endif

// ---- kernel 0: split W_base (f32) into hi/lo bf16 --------------------------
__global__ void wsplit_kernel(const float* __restrict__ W,
                              bf16_t* __restrict__ Whi,
                              bf16_t* __restrict__ Wlo) {
  int idx = (blockIdx.x * 256 + threadIdx.x) * 4;   // 1024 blocks cover 1M elems
  f32x4 w = *(const f32x4*)(W + idx);
#pragma unroll
  for (int e = 0; e < 4; ++e) {
    float f  = w[e];
    bf16_t h = (bf16_t)f;
    Whi[idx + e] = h;
    Wlo[idx + e] = (bf16_t)(f - (float)h);
  }
}

// ---- kernel 1: hypernetwork -> A (bf16, padded to 16 rows) and B_mat (f32) -
__global__ void hyper_kernel(const float* __restrict__ style,
                             const float* __restrict__ W1,
                             const float* __restrict__ b1,
                             const float* __restrict__ W2,
                             const float* __restrict__ b2,
                             bf16_t* __restrict__ Abf,   // [B][16][IN_F]
                             float*  __restrict__ Bm) {  // [B][OUT_F][RANK]
  const int b = blockIdx.x;
  const int t = threadIdx.x;
  __shared__ float ss[STYLE];
  __shared__ float sh[STYLE / 2];

  ss[t] = style[b * STYLE + t];
  __syncthreads();

  if (t < STYLE / 2) {
    float a = b1[t];
    const float* w = W1 + t * STYLE;
#pragma unroll 4
    for (int j = 0; j < STYLE; ++j) a += w[j] * ss[j];
    sh[t] = a / (1.f + __expf(-a));           // silu
  }
  __syncthreads();

  const int total = (IN_F + OUT_F) * RANK;    // 16384
#pragma unroll 1
  for (int i = 0; i < total / 256; ++i) {
    int idx = i * 256 + t;
    float a = b2[idx];
    const float* w = W2 + (size_t)idx * (STYLE / 2);
#pragma unroll 4
    for (int j = 0; j < STYLE / 2; ++j) a += w[j] * sh[j];
    if (idx < RANK * IN_F) {                  // A part: [r][k]
      int r = idx >> 10, k = idx & (IN_F - 1);
      Abf[((b * 16 + r) << 10) + k] = (bf16_t)a;
    } else {                                  // B part: [o][r]
      int j2 = idx - RANK * IN_F;
      Bm[(size_t)b * OUT_F * RANK + j2] = a;
    }
  }
  // zero the 8 pad rows of A so the 16-wide WMMA B operand is exact
#pragma unroll 1
  for (int i = 0; i < 32; ++i) {
    int p = i * 256 + t;                      // p < 8192
    Abf[((b * 16 + 8 + (p >> 10)) << 10) + (p & (IN_F - 1))] = (bf16_t)0.f;
  }
}

// ---- kernel 2: fused base GEMM + LoRA, bf16 hi/lo split WMMA ---------------
// LDS arena layout (bytes):
//   sxh      : [     0,  8192)   x hi   (single buffer, 128x32 bf16)
//   sxl      : [  8192, 16384)   x lo
//   swh x2   : [ 16384, 32768)   W hi   (double buffer, 8KB each)
//   swl x2   : [ 32768, 49152)   W lo   (double buffer, 8KB each)
//   sa  x2   : [ 49152, 51200)   LoRA A (double buffer, 1KB each)
//   s_lr     : [ 51200, 55296)   low_rank spill (128x8 f32)
//   s_bm     : [ 55296, 59392)   B_mat tile     (128x8 f32)
//   s_bb     : [ 59392, 59904)   bias tile      (128 f32)
#define SMEM_BYTES 59904

__global__ __launch_bounds__(256)
void stylelinear_kernel(const float*  __restrict__ x,
                        const bf16_t* __restrict__ Whi,
                        const bf16_t* __restrict__ Wlo,
                        const float*  __restrict__ b_base,
                        const bf16_t* __restrict__ Abf,
                        const float*  __restrict__ Bm,
                        float*        __restrict__ out) {
  __shared__ __align__(16) unsigned char smem[SMEM_BYTES];
  bf16_t* const sxh  = (bf16_t*)(smem);
  bf16_t* const sxl  = (bf16_t*)(smem + 8192);
  bf16_t* const swh0 = (bf16_t*)(smem + 16384);   // + buf*4096 elems
  bf16_t* const swl0 = (bf16_t*)(smem + 32768);   // + buf*4096 elems
  bf16_t* const sa0  = (bf16_t*)(smem + 49152);   // + buf*512  elems
  float*  const s_lr = (float*)(smem + 51200);
  float*  const s_bm = (float*)(smem + 55296);
  float*  const s_bb = (float*)(smem + 59392);
#define SWH(buf) (swh0 + (buf) * 4096)
#define SWL(buf) (swl0 + (buf) * 4096)
#define SA(buf)  (sa0  + (buf) * 512)

  const int t    = threadIdx.x;
  const int lane = t & 31;
  const int wave = t >> 5;
  const int m0   = blockIdx.y * 128;          // global row tile (within one batch)
  const int n0   = blockIdx.x * 128;          // output-feature tile
  const int b    = blockIdx.y >> 5;           // 4096/128 = 32 row tiles per batch

  if (t < 128) s_bb[t] = b_base[n0 + t];

  v8f acc[8];
#pragma unroll
  for (int i = 0; i < 8; ++i) acc[i] = (v8f){0.f,0.f,0.f,0.f,0.f,0.f,0.f,0.f};
  v8f accLR = (v8f){0.f,0.f,0.f,0.f,0.f,0.f,0.f,0.f};

  // fragment addressing (wave32, 16-bit operands), contiguous k order shared by A and B
  const int arow  = lane & 15;                // A: row; K contiguous 0..15 / 16..31
  const int akoff = (lane < 16) ? 0 : 16;
  const int bcol  = lane & 15;                // B: col; K contiguous 0..15 / 16..31
  const int bkoff = (lane < 16) ? 0 : 16;
  const int rm    = wave * 16;                // this wave's 16-row slab

  // per-thread staging coordinates
  const int w_col0 = t >> 2;                  // W chunk 0: col, 8-half k offset
  const int w_kq   = (t & 3) * 8;
  const int a_r    = t >> 4;                  // A: 1 b32 (2 halves) per thread
  const int a_k    = (t & 15) * 2;

#if ASYNC_OK
  // ---- issue async W/A tile copies for k0 into buffer `buf` (5 per wave) ----
  #define ISSUE_WA(k_next, buf)                                                   \
    do {                                                                          \
      __builtin_amdgcn_global_load_async_to_lds_b32(                              \
          (as1_i32*)(Abf + ((b * 16 + a_r) << 10) + (k_next) + a_k),              \
          (as3_i32*)(SA(buf) + a_r * 32 + a_k), 0, 0);                            \
      _Pragma("unroll")                                                           \
      for (int i = 0; i < 2; ++i) {                                               \
        int col = w_col0 + i * 64;                                                \
        size_t g = (size_t)(n0 + col) * IN_F + (k_next) + w_kq;                   \
        __builtin_amdgcn_global_load_async_to_lds_b128(                           \
            (as1_i32x4*)(Whi + g),                                                \
            (as3_i32x4*)(SWH(buf) + col * 32 + w_kq), 0, 0);                      \
        __builtin_amdgcn_global_load_async_to_lds_b128(                           \
            (as1_i32x4*)(Wlo + g),                                                \
            (as3_i32x4*)(SWL(buf) + col * 32 + w_kq), 0, 0);                      \
      }                                                                           \
    } while (0)
  ISSUE_WA(0, 0);
#endif

  // preload first x tile into registers (4 x float4 per thread)
  f32x4 xr[4];
#pragma unroll
  for (int i = 0; i < 4; ++i) {
    int lin = i * 256 + t;
    xr[i] = *(const f32x4*)(x + (size_t)(m0 + (lin >> 3)) * IN_F + (lin & 7) * 4);
  }

  for (int k0 = 0; k0 < IN_F; k0 += 32) {
    const int cur = (k0 >> 5) & 1;
    const bool has_next = (k0 + 32) < IN_F;

#if ASYNC_OK
    if (has_next) ISSUE_WA(k0 + 32, cur ^ 1);
#endif

    // ---- stage x tile from registers: f32 -> hi/lo bf16 into LDS ----
#pragma unroll
    for (int i = 0; i < 4; ++i) {
      int lin = i * 256 + t;
      int row = lin >> 3;
      int kq  = (lin & 7) * 4;
#pragma unroll
      for (int e = 0; e < 4; ++e) {
        float f  = xr[i][e];
        bf16_t h = (bf16_t)f;
        sxh[row * 32 + kq + e] = h;
        sxl[row * 32 + kq + e] = (bf16_t)(f - (float)h);
      }
    }
    // ---- preload next x tile into registers (overlaps with compute) ----
    if (has_next) {
#pragma unroll
      for (int i = 0; i < 4; ++i) {
        int lin = i * 256 + t;
        xr[i] = *(const f32x4*)(x + (size_t)(m0 + (lin >> 3)) * IN_F +
                                (k0 + 32) + (lin & 7) * 4);
      }
      // speculative prefetch two tiles ahead (global_prefetch_b8)
      if (k0 + 64 < IN_F)
        __builtin_prefetch(x + (size_t)(m0 + (t >> 1)) * IN_F + (k0 + 64) +
                               (t & 1) * 16, 0, 3);
    }

#if ASYNC_OK
    // retire current tile's async copies (next tile's 5 may stay in flight)
    if (has_next) __builtin_amdgcn_s_wait_asynccnt(5);
    else          __builtin_amdgcn_s_wait_asynccnt(0);
#else
    // ---- fallback: synchronous W/A staging ----
#pragma unroll
    for (int i = 0; i < 2; ++i) {
      int col = w_col0 + i * 64;
      size_t g = (size_t)(n0 + col) * IN_F + k0 + w_kq;
      *(u32x4*)(SWH(cur) + col * 32 + w_kq) = *(const u32x4*)(Whi + g);
      *(u32x4*)(SWL(cur) + col * 32 + w_kq) = *(const u32x4*)(Wlo + g);
    }
    if (t < 64) {
      int r  = t >> 2;
      int kq = (t & 3) * 8;
      *(u32x4*)(SA(cur) + r * 32 + kq) =
          *(const u32x4*)(Abf + ((b * 16 + r) << 10) + k0 + kq);
    }
#endif
    __syncthreads();

    // ---- A fragments for this wave's rows ----
    union frag { v16bf v; v8bf h[2]; };
    frag uxh, uxl;
    uxh.h[0] = *(const v8bf*)(sxh + (rm + arow) * 32 + akoff);
    uxh.h[1] = *(const v8bf*)(sxh + (rm + arow) * 32 + akoff + 8);
    uxl.h[0] = *(const v8bf*)(sxl + (rm + arow) * 32 + akoff);
    uxl.h[1] = *(const v8bf*)(sxl + (rm + arow) * 32 + akoff + 8);

    // ---- LoRA low_rank accumulation (x_hi @ A^T), rank 8 padded to 16 ----
    {
      frag ua;
      ua.h[0] = *(const v8bf*)(SA(cur) + bcol * 32 + bkoff);
      ua.h[1] = *(const v8bf*)(SA(cur) + bcol * 32 + bkoff + 8);
      accLR = __builtin_amdgcn_wmma_f32_16x16x32_bf16(
                  false, uxh.v, false, ua.v, (short)0, accLR, false, false);
    }

    // ---- main GEMM: 8 column subtiles x 3 split-precision WMMAs ----
#pragma unroll
    for (int c = 0; c < 8; ++c) {
      frag uh, ul;
      uh.h[0] = *(const v8bf*)(SWH(cur) + (c * 16 + bcol) * 32 + bkoff);
      uh.h[1] = *(const v8bf*)(SWH(cur) + (c * 16 + bcol) * 32 + bkoff + 8);
      ul.h[0] = *(const v8bf*)(SWL(cur) + (c * 16 + bcol) * 32 + bkoff);
      ul.h[1] = *(const v8bf*)(SWL(cur) + (c * 16 + bcol) * 32 + bkoff + 8);
      acc[c] = __builtin_amdgcn_wmma_f32_16x16x32_bf16(
                   false, uxh.v, false, uh.v, (short)0, acc[c], false, false);
      acc[c] = __builtin_amdgcn_wmma_f32_16x16x32_bf16(
                   false, uxh.v, false, ul.v, (short)0, acc[c], false, false);
      acc[c] = __builtin_amdgcn_wmma_f32_16x16x32_bf16(
                   false, uxl.v, false, uh.v, (short)0, acc[c], false, false);
    }
    __syncthreads();   // protect buffers before next iteration's writes
  }

  // ---- spill low_rank (C/D layout: vgpr v, lane l -> row (l>=16?8:0)+v, col l&15)
  {
    int rowb = rm + ((lane >> 4) << 3);
    int col  = lane & 15;
    if (col < RANK) {
#pragma unroll
      for (int v = 0; v < 8; ++v) s_lr[(rowb + v) * RANK + col] = accLR[v];
    }
  }
  // ---- stage B_mat tile [128 cols][8] ----
#pragma unroll
  for (int i = 0; i < 4; ++i) {
    int lin = i * 256 + t;                    // < 1024
    s_bm[lin] = Bm[(size_t)b * OUT_F * RANK + (size_t)n0 * RANK + lin];
  }
  __syncthreads();

  // ---- epilogue: out = acc + bias + low_rank @ Bm^T ----
  {
    int rowb = rm + ((lane >> 4) << 3);
#pragma unroll
    for (int c = 0; c < 8; ++c) {
      int col  = c * 16 + (lane & 15);
      float bb = s_bb[col];
#pragma unroll
      for (int v = 0; v < 8; ++v) {
        int row   = rowb + v;
        float val = acc[c][v] + bb;
#pragma unroll
        for (int r = 0; r < RANK; ++r)
          val += s_lr[row * RANK + r] * s_bm[col * RANK + r];
        out[(size_t)(m0 + row) * OUT_F + (n0 + col)] = val;
      }
    }
  }
}

// ---- host launcher ---------------------------------------------------------
extern "C" void kernel_launch(void* const* d_in, const int* in_sizes, int n_in,
                              void* d_out, int out_size, void* d_ws, size_t ws_size,
                              hipStream_t stream) {
  (void)in_sizes; (void)n_in; (void)out_size; (void)ws_size;
  const float* x     = (const float*)d_in[0];
  const float* style = (const float*)d_in[1];
  const float* Wb    = (const float*)d_in[2];
  const float* bb    = (const float*)d_in[3];
  const float* W1    = (const float*)d_in[4];
  const float* b1    = (const float*)d_in[5];
  const float* W2    = (const float*)d_in[6];
  const float* b2    = (const float*)d_in[7];
  float* out = (float*)d_out;

  char* ws = (char*)d_ws;                       // 5 MB total scratch
  bf16_t* Abf = (bf16_t*)(ws);                  // 512 KB : [16][16][1024] bf16
  float*  Bm  = (float*) (ws + (512u << 10));   // 512 KB : [16][1024][8] f32
  bf16_t* Whi = (bf16_t*)(ws + (1024u << 10));  // 2 MB   : bf16 hi
  bf16_t* Wlo = (bf16_t*)(ws + (3072u << 10));  // 2 MB   : bf16 lo

  wsplit_kernel<<<1024, 256, 0, stream>>>(Wb, Whi, Wlo);
  hyper_kernel<<<NBATCH, 256, 0, stream>>>(style, W1, b1, W2, b2, Abf, Bm);
  stylelinear_kernel<<<dim3(OUT_F / 128, (NBATCH * NSEQ) / 128), 256, 0, stream>>>(
      x, Whi, Wlo, bb, Abf, Bm, out);
}
